// DependencyLearner_3367254360621
// MI455X (gfx1250) — matrix operand order repaired and checked by reading, not compiled, not run on hardware
//
#include <hip/hip_runtime.h>
#include <hip/hip_bf16.h>
#include <math.h>

#define VOCAB_N 100000
#define Dd 256
#define Bb 64
#define Ll 128

typedef _Float16 halfT;
typedef _Float16 v16h __attribute__((ext_vector_type(16)));
typedef float    v8f  __attribute__((ext_vector_type(8)));
typedef float    f32x4 __attribute__((ext_vector_type(4)));

// ---------------------------------------------------------------------------
// threefry2x32 (exact JAX PRNG), key = (0, 123) from jax.random.key(123)
// ---------------------------------------------------------------------------
__device__ __forceinline__ unsigned rotl32(unsigned x, int n) {
    return (x << n) | (x >> (32 - n));
}

__device__ __forceinline__ void threefry2x32(unsigned k0, unsigned k1,
                                             unsigned x0, unsigned x1,
                                             unsigned* o0, unsigned* o1) {
    const unsigned ks2 = k0 ^ k1 ^ 0x1BD11BDAu;
    x0 += k0; x1 += k1;
    #pragma unroll
    for (int i = 0; i < 4; ++i) { const int r[4] = {13,15,26,6};  x0 += x1; x1 = rotl32(x1, r[i]); x1 ^= x0; }
    x0 += k1; x1 += ks2 + 1u;
    #pragma unroll
    for (int i = 0; i < 4; ++i) { const int r[4] = {17,29,16,24}; x0 += x1; x1 = rotl32(x1, r[i]); x1 ^= x0; }
    x0 += ks2; x1 += k0 + 2u;
    #pragma unroll
    for (int i = 0; i < 4; ++i) { const int r[4] = {13,15,26,6};  x0 += x1; x1 = rotl32(x1, r[i]); x1 ^= x0; }
    x0 += k0; x1 += k1 + 3u;
    #pragma unroll
    for (int i = 0; i < 4; ++i) { const int r[4] = {17,29,16,24}; x0 += x1; x1 = rotl32(x1, r[i]); x1 ^= x0; }
    x0 += k1; x1 += ks2 + 4u;
    #pragma unroll
    for (int i = 0; i < 4; ++i) { const int r[4] = {13,15,26,6};  x0 += x1; x1 = rotl32(x1, r[i]); x1 ^= x0; }
    x0 += ks2; x1 += k0 + 5u;
    *o0 = x0; *o1 = x1;
}

// ---------------------------------------------------------------------------
// Kernel 1: gather masked embeddings into fp16 tiles
//   vecH[b,l,:]   = (f16) V[words_masked[b,l], :]
//   covecH[b,l,:] = (f16) W[words_masked[b,l], :]
// ---------------------------------------------------------------------------
__global__ void gather_f16_kernel(const int* __restrict__ ps,
                                  const int* __restrict__ mask,
                                  const float* __restrict__ V,
                                  const float* __restrict__ W,
                                  halfT* __restrict__ vecH,
                                  halfT* __restrict__ covecH) {
    const int bl = blockIdx.x;          // b*L + l
    const int b  = bl / Ll;
    const int l  = bl % Ll;
    const int d  = threadIdx.x;
    const bool pad = mask[bl] != 0;
    const int  w   = pad ? 0 : ps[b * 3 * Ll + l];
    vecH  [(size_t)bl * Dd + d] = (halfT)V[(size_t)w * Dd + d];
    covecH[(size_t)bl * Dd + d] = (halfT)W[(size_t)w * Dd + d];
}

// ---------------------------------------------------------------------------
// Kernel 2: energies[b] = covec[b] (128x256) x vec[b]^T (256x128), f32 accum.
// One wave32 per 16x64 output strip, double-buffered fragments:
// K-step k+1's A + 4 B fragments are prefetched into an independent register
// set while the 4 WMMAs of step k issue -> staggered s_wait_loadcnt, no
// exposed load latency. 32x v_wmma_f32_16x16x32_f16 per wave.
// Fragment layouts follow CDNA5 ISA §7.12.2 (16-bit A/B, wave32).
// ---------------------------------------------------------------------------
__global__ void energies_wmma_kernel(const halfT* __restrict__ covecH,
                                     const halfT* __restrict__ vecH,
                                     float* __restrict__ E) {
    const int lane  = threadIdx.x;        // 0..31
    const int b     = blockIdx.x >> 4;    // 16 strips per batch
    const int strip = blockIdx.x & 15;
    const int tm    = strip >> 1;         // 0..7  row tile (16 rows)
    const int tn4   = strip & 1;          // 0..1  column strip (64 cols)
    const bool hi   = lane >= 16;
    const int  lr   = lane & 15;

    const halfT* Arow  = covecH + ((size_t)b * Ll + tm * 16 + lr) * Dd;  // M row
    const halfT* Brow0 = vecH   + ((size_t)b * Ll + tn4 * 64 + lr) * Dd; // N col j*16+lr

    union Frag { v16h v; f32x4 q[2]; };
    v8f acc[4] = {v8f{}, v8f{}, v8f{}, v8f{}};

    // A 16x32: lanes 0-15 -> K {k0+0..7, k0+16..23}; lanes 16-31 -> K {k0+8..15, k0+24..31}
    const int dka = hi ? 8 : 0;
    // B 32x16: lanes 0-15 -> K {k0+0..7, k0+16..23}; lanes 16-31 -> K {k0+8..15, k0+24..31}
    const int dkb = hi ? 8 : 0;

    Frag a_cur, a_nxt;
    Frag b_cur[4], b_nxt[4];

    // prologue: load K-step 0
    a_cur.q[0] = *(const f32x4*)(Arow + dka);
    a_cur.q[1] = *(const f32x4*)(Arow + dka + 16);
    #pragma unroll
    for (int j = 0; j < 4; ++j) {
        const halfT* Brow = Brow0 + (size_t)j * 16 * Dd;
        b_cur[j].q[0] = *(const f32x4*)(Brow + dkb);
        b_cur[j].q[1] = *(const f32x4*)(Brow + dkb + 16);
    }

    #pragma unroll
    for (int k0 = 0; k0 < Dd; k0 += 32) {
        if (k0 + 32 < Dd) {
            const int ka = k0 + 32 + dka;
            const int kb = k0 + 32 + dkb;
            a_nxt.q[0] = *(const f32x4*)(Arow + ka);
            a_nxt.q[1] = *(const f32x4*)(Arow + ka + 16);
            #pragma unroll
            for (int j = 0; j < 4; ++j) {
                const halfT* Brow = Brow0 + (size_t)j * 16 * Dd;
                b_nxt[j].q[0] = *(const f32x4*)(Brow + kb);
                b_nxt[j].q[1] = *(const f32x4*)(Brow + kb + 16);
            }
        }
        #pragma unroll
        for (int j = 0; j < 4; ++j) {
            acc[j] = __builtin_amdgcn_wmma_f32_16x16x32_f16(
                /*neg_a=*/false, a_cur.v, /*neg_b=*/false, b_cur[j].v,
                /*c_mod=*/(short)0, acc[j], /*reuse_a=*/false, /*reuse_b=*/false);
        }
        a_cur = a_nxt;
        #pragma unroll
        for (int j = 0; j < 4; ++j) b_cur[j] = b_nxt[j];
    }

    // C/D layout: VGPR r -> (M = r + (hi?8:0), N = lane&15) within each tile
    float* Eb = E + (size_t)b * Ll * Ll;
    #pragma unroll
    for (int j = 0; j < 4; ++j) {
        #pragma unroll
        for (int r = 0; r < 8; ++r) {
            const int m = tm * 16 + r + (hi ? 8 : 0);
            const int n = tn4 * 64 + j * 16 + lr;
            Eb[m * Ll + n] = acc[j][r];
        }
    }
}

// ---------------------------------------------------------------------------
// Kernel 3: categorical sampling (Gumbel-argmax, JAX threefry semantics).
// One block (128 threads) per (b, l) row.
// ---------------------------------------------------------------------------
__global__ void sample_kernel(const float* __restrict__ E,
                              const int* __restrict__ mask,
                              int* __restrict__ negHeads) {
    const int b = blockIdx.x / Ll;
    const int l = blockIdx.x % Ll;
    const int m = threadIdx.x;

    const bool rowPad = mask[b * Ll + l] != 0;
    if (rowPad) {                       // sample overwritten with 0 for pads
        if (m == 0) negHeads[b * Ll + l] = 0;
        return;
    }

    const bool colPad = mask[b * Ll + m] != 0;
    float logit;
    if (colPad || m == l) logit = -__builtin_inff();
    else                  logit = E[(size_t)b * Ll * Ll + l * Ll + m];

    // JAX uniform bits: counter = flat index over (B,L,L); iota split into halves
    const unsigned N    = (unsigned)(Bb * Ll * Ll);
    const unsigned half = N / 2u;
    const unsigned idx  = (unsigned)((b * Ll + l) * Ll + m);
    unsigned x0, x1, o0, o1;
    if (idx < half) { x0 = idx;        x1 = idx + half; }
    else            { x0 = idx - half; x1 = idx;        }
    threefry2x32(0u, 123u, x0, x1, &o0, &o1);
    const unsigned bits = (idx < half) ? o0 : o1;

    float f = __uint_as_float((bits >> 9) | 0x3f800000u) - 1.0f;   // [0,1)
    const float tiny = 1.17549435e-38f;
    float u = f * (1.0f - tiny) + tiny;
    u = fmaxf(u, tiny);
    const float g = -logf(-logf(u));
    const float val = logit + g;       // -inf stays -inf

    __shared__ float sv[Ll];
    __shared__ int   si[Ll];
    sv[m] = val; si[m] = m;
    __syncthreads();
    for (int o = Ll / 2; o > 0; o >>= 1) {
        if (m < o) {
            const float v2 = sv[m + o]; const int i2 = si[m + o];
            if (v2 > sv[m] || (v2 == sv[m] && i2 < si[m])) { sv[m] = v2; si[m] = i2; }
        }
        __syncthreads();
    }
    if (m == 0) negHeads[b * Ll + l] = si[0];
}

// ---------------------------------------------------------------------------
// Kernel 4: per-batch masked score sums (positive: blocks [0,B), negative: [B,2B))
// ---------------------------------------------------------------------------
__global__ void score_kernel(const int* __restrict__ ps,
                             const int* __restrict__ mask,
                             const float* __restrict__ V,
                             const float* __restrict__ W,
                             const float* __restrict__ vb,
                             const float* __restrict__ wb,
                             const int* __restrict__ negHeads,
                             float* __restrict__ out) {
    const bool positive = blockIdx.x < Bb;
    const int  b = blockIdx.x % Bb;
    const int  l = threadIdx.x;

    const int* words = ps + b * 3 * Ll;        // positive_sentences[:,0,:]
    const int* hids  = ps + b * 3 * Ll + Ll;   // positive_sentences[:,1,:]

    const bool pad  = mask[b * Ll + l] != 0;
    const bool incl = pad || (l == 0);         // mask_incl_root

    float sc = 0.0f;
    if (!incl) {
        const int w = words[l];                                // not padded here
        const int h = positive ? hids[l] : negHeads[b * Ll + l];
        const bool padh = mask[b * Ll + h] != 0;
        const int  hw   = padh ? 0 : words[h];                 // heads via masked words
        float s = 0.0f;
        #pragma unroll 4
        for (int d = 0; d < Dd; ++d)
            s = fmaf(W[(size_t)w * Dd + d], V[(size_t)hw * Dd + d], s);
        sc = s + vb[hw] + wb[w];
    }

    __shared__ float red[Ll];
    red[l] = sc;
    __syncthreads();
    for (int o = Ll / 2; o > 0; o >>= 1) {
        if (l < o) red[l] += red[l + o];
        __syncthreads();
    }
    if (l == 0) out[(positive ? 0 : Bb) + b] = red[0];
}

// ---------------------------------------------------------------------------
extern "C" void kernel_launch(void* const* d_in, const int* in_sizes, int n_in,
                              void* d_out, int out_size, void* d_ws, size_t ws_size,
                              hipStream_t stream) {
    (void)in_sizes; (void)n_in; (void)out_size; (void)ws_size;
    // inputs (setup_inputs order): batch_id, positive_sentences, mask, V, W, vb, wb
    const int*   ps   = (const int*)d_in[1];    // (B,3,L)
    const int*   mask = (const int*)d_in[2];    // (B,L) bool->int
    const float* V    = (const float*)d_in[3];  // (VOCAB,D)
    const float* W    = (const float*)d_in[4];  // (COVOCAB,D)
    const float* vb   = (const float*)d_in[5];  // (VOCAB,)
    const float* wb   = (const float*)d_in[6];  // (COVOCAB,)
    float* out = (float*)d_out;                 // [0:B) positive, [B:2B) negative

    // workspace carve-up (~12.03 MB)
    halfT* vecH   = (halfT*)d_ws;                        // B*L*D f16  (4 MB)
    halfT* covecH = vecH + (size_t)Bb * Ll * Dd;         // B*L*D f16  (4 MB)
    float* E      = (float*)(covecH + (size_t)Bb * Ll * Dd); // B*L*L f32 (4 MB)
    int*   negH   = (int*)(E + (size_t)Bb * Ll * Ll);    // B*L int    (32 KB)

    gather_f16_kernel   <<<Bb * Ll, Dd,  0, stream>>>(ps, mask, V, W, vecH, covecH);
    energies_wmma_kernel<<<Bb * 16, 32,  0, stream>>>(covecH, vecH, E);
    sample_kernel       <<<Bb * Ll, Ll,  0, stream>>>(E, mask, negH);
    score_kernel        <<<2 * Bb,  Ll,  0, stream>>>(ps, mask, V, W, vb, wb, negH, out);
}
